// GNNens_85581518340494
// MI455X (gfx1250) — matrix operand order repaired and checked by reading, not compiled, not run on hardware
//
#include <hip/hip_runtime.h>

// ---------------------------------------------------------------------------
// GNNens routed mixture-of-MLPs for gfx1250 (MI455X).
//   out[n,:] = (sigmoid(lat @ W1[c] + b1[c]) @ W2[c] + b2[c]),  c = comp_idx[n]
//   lat = eps[n,:] * softplus(rho[c]) + mu[c,:]
// Strategy: bucket samples by component, then per-component fused 2-layer
// GEMM with bf16 WMMA (f32 accumulate), h kept entirely in LDS.
// ---------------------------------------------------------------------------

typedef __attribute__((ext_vector_type(16))) __bf16 v16bf;
typedef __attribute__((ext_vector_type(8)))  __bf16 v8bf;
typedef __attribute__((ext_vector_type(8)))  float  v8f;

#define NB_COMP   8
#define LAT_DIM   512
#define NB_NEUR   1024
#define OUT_DIM   512
#define N_SAMPLES 16384
#define M_TILE    64
#define MAX_TILES (N_SAMPLES / M_TILE + NB_COMP)   // 264

// ---------------- workspace layout (bytes) ----------------
#define WS_COUNTS   0          // 8 ints
#define WS_OFFSETS  64         // 8 ints
#define WS_ORDER    128        // 16384 ints  (64 KB)
#define WS_B1PACK   (128 + 4 * N_SAMPLES)                       // 8 MB bf16
#define WS_B2PACK   (WS_B1PACK + 2 * NB_COMP * LAT_DIM * NB_NEUR) // 8 MB bf16

// ---------------------------------------------------------------------------
// 1) Routing: histogram + prefix + scatter, single 1024-thread block.
// ---------------------------------------------------------------------------
__global__ void route_kernel(const int* __restrict__ comp,
                             int* __restrict__ counts_g,
                             int* __restrict__ offsets_g,
                             int* __restrict__ order) {
  __shared__ int cnt[NB_COMP];
  __shared__ int cur[NB_COMP];
  __shared__ int off[NB_COMP];
  const int tid = threadIdx.x;
  if (tid < NB_COMP) { cnt[tid] = 0; cur[tid] = 0; }
  __syncthreads();
  for (int n = tid; n < N_SAMPLES; n += 1024)
    atomicAdd(&cnt[comp[n] & (NB_COMP - 1)], 1);
  __syncthreads();
  if (tid == 0) {
    int run = 0;
    for (int c = 0; c < NB_COMP; ++c) { off[c] = run; run += cnt[c]; }
  }
  __syncthreads();
  for (int n = tid; n < N_SAMPLES; n += 1024) {
    int c = comp[n] & (NB_COMP - 1);
    int p = atomicAdd(&cur[c], 1);
    order[off[c] + p] = n;
  }
  if (tid < NB_COMP) { counts_g[tid] = cnt[tid]; offsets_g[tid] = off[tid]; }
}

// ---------------------------------------------------------------------------
// 2) Weight pre-pack: f32 row-major [C][K][N] -> bf16 WMMA B-fragments.
//    Fragment (c,kt,nt): 32 lanes x 16 bf16.  Lane l<16 -> col n0+l,
//    K = {k0..k0+7} U {k0+16..k0+23}; lane l>=16 -> col n0+(l&15),
//    K = {k0+8..k0+15} U {k0+24..k0+31}.  (ISA 05_wmma 7.12.2, 16-bit B layout.)
// ---------------------------------------------------------------------------
__global__ void pack_kernel(const float* __restrict__ W,
                            __bf16* __restrict__ out,
                            int KT, int NT, int Ncols) {
  const int gid   = blockIdx.x * blockDim.x + threadIdx.x;
  const int total = NB_COMP * KT * NT * 32;
  if (gid >= total) return;
  const int lane = gid & 31;
  const int frag = gid >> 5;
  const int nt = frag % NT;
  const int kt = (frag / NT) % KT;
  const int c  = frag / (NT * KT);
  const int n     = nt * 16 + (lane & 15);
  const int kbase = kt * 32 + ((lane < 16) ? 0 : 8);
  const size_t K  = (size_t)KT * 32;
  const float* Wc = W + (size_t)c * K * Ncols;
  v16bf v;
#pragma unroll
  for (int j = 0; j < 16; ++j)
    v[j] = (__bf16)Wc[(size_t)(kbase + (j < 8 ? j : j + 8)) * Ncols + n];
  *reinterpret_cast<v16bf*>(out + ((size_t)frag << 9) + (lane << 4)) = v;
}

// ---------------------------------------------------------------------------
// A-fragment load from LDS row-major bf16 (stride elems): two 16B chunks/lane.
// Lane l<16 -> row m0+l, K {k0..k0+7} U {k0+16..k0+23}; lanes 16..31 -> +8.
// ---------------------------------------------------------------------------
__device__ __forceinline__ v16bf load_a_frag(const __bf16* base, int m0,
                                             int k0, int stride, int lane) {
  const int l  = lane & 15;
  const int hi = (lane >> 4) & 1;
  const __bf16* p = base + (size_t)(m0 + l) * stride + k0 + hi * 8;
  v8bf lo = *reinterpret_cast<const v8bf*>(p);
  v8bf hc = *reinterpret_cast<const v8bf*>(p + 16);
  return __builtin_shufflevector(lo, hc, 0, 1, 2, 3, 4, 5, 6, 7,
                                 8, 9, 10, 11, 12, 13, 14, 15);
}

// ---------------------------------------------------------------------------
// 3) Fused per-component 2-layer MLP. One block = 64 gathered rows of one
//    component. 256 threads = 8 waves. Dynamic LDS:
//      latA  64x512  bf16 =  64 KB
//      hbuf  64x1024 bf16 = 128 KB
//      srow  64 int       = 256 B
// ---------------------------------------------------------------------------
__global__ __launch_bounds__(256) void mlp_kernel(
    const float* __restrict__ eps, const int* __restrict__ order,
    const int* __restrict__ counts, const int* __restrict__ offsets,
    const float* __restrict__ mu, const float* __restrict__ rho,
    const float* __restrict__ b1, const float* __restrict__ b2,
    const __bf16* __restrict__ B1p, const __bf16* __restrict__ B2p,
    float* __restrict__ out) {
  extern __shared__ char smem_raw[];
  __bf16* latA = reinterpret_cast<__bf16*>(smem_raw);
  __bf16* hbuf = reinterpret_cast<__bf16*>(smem_raw + M_TILE * LAT_DIM * 2);
  int* srow = reinterpret_cast<int*>(smem_raw + M_TILE * LAT_DIM * 2 +
                                     M_TILE * NB_NEUR * 2);

  // -------- tile mapping (uniform across block) --------
  int c = -1, t = 0, run = 0;
  for (int cc = 0; cc < NB_COMP; ++cc) {
    int tiles = (counts[cc] + M_TILE - 1) / M_TILE;
    if (c < 0 && (int)blockIdx.x < run + tiles) { c = cc; t = blockIdx.x - run; }
    run += tiles;
  }
  if (c < 0) return;  // block-uniform exit
  const int cnt  = counts[c];
  const int goff = offsets[c] + t * M_TILE;
  const int rows = min(M_TILE, cnt - t * M_TILE);

  const int tid  = threadIdx.x;
  const int wave = tid >> 5;
  const int lane = tid & 31;

  // softplus(rho[c]) (numerically stable)
  const float rc    = rho[c];
  const float sigma = (rc > 20.f) ? rc : log1pf(expf(rc));

  if (tid < M_TILE)
    srow[tid] = order[goff + ((tid < rows) ? tid : 0)];
  __syncthreads();

  // -------- build lat tile in LDS (bf16) --------
  const float* mu_c = mu + (size_t)c * LAT_DIM;
  for (int e = tid; e < M_TILE * LAT_DIM; e += 256) {
    int m = e >> 9;          // /512
    int d = e & (LAT_DIM - 1);
    float v = eps[(size_t)srow[m] * LAT_DIM + d] * sigma + mu_c[d];
    latA[(size_t)m * LAT_DIM + d] = (__bf16)v;
  }
  __syncthreads();

  const float* b1c = b1 + (size_t)c * NB_NEUR;
  const float* b2c = b2 + (size_t)c * OUT_DIM;
  const int nlane = lane & 15;
  const int mlane = (lane < 16) ? 0 : 8;

  // -------- layer 1: h = sigmoid(lat @ W1 + b1), each wave: 128 columns ----
#pragma unroll 1
  for (int i = 0; i < 8; ++i) {
    const int nt = wave * 8 + i;          // column tile 0..63
    const int n0 = nt * 16;
    const float bias = b1c[n0 + nlane];
    v8f acc0 = {}, acc1 = {}, acc2 = {}, acc3 = {};
#pragma unroll
    for (int k0 = 0; k0 < LAT_DIM; k0 += 32) {
      const size_t frag = ((size_t)(c * 16 + (k0 >> 5)) * 64 + nt);
      v16bf b = *reinterpret_cast<const v16bf*>(B1p + (frag << 9) + (lane << 4));
      v16bf a0 = load_a_frag(latA, 0, k0, LAT_DIM, lane);
      v16bf a1 = load_a_frag(latA, 16, k0, LAT_DIM, lane);
      v16bf a2 = load_a_frag(latA, 32, k0, LAT_DIM, lane);
      v16bf a3 = load_a_frag(latA, 48, k0, LAT_DIM, lane);
      acc0 = __builtin_amdgcn_wmma_f32_16x16x32_bf16(false, a0, false, b, (short)0, acc0, false, false);
      acc1 = __builtin_amdgcn_wmma_f32_16x16x32_bf16(false, a1, false, b, (short)0, acc1, false, false);
      acc2 = __builtin_amdgcn_wmma_f32_16x16x32_bf16(false, a2, false, b, (short)0, acc2, false, false);
      acc3 = __builtin_amdgcn_wmma_f32_16x16x32_bf16(false, a3, false, b, (short)0, acc3, false, false);
    }
    // sigmoid + scatter into hbuf (bf16)
    v8f* accs[4] = {&acc0, &acc1, &acc2, &acc3};
#pragma unroll
    for (int m = 0; m < 4; ++m) {
#pragma unroll
      for (int r = 0; r < 8; ++r) {
        int row = m * 16 + r + mlane;
        float v = (*accs[m])[r] + bias;
        v = 1.0f / (1.0f + expf(-v));
        hbuf[(size_t)row * NB_NEUR + n0 + nlane] = (__bf16)v;
      }
    }
  }
  __syncthreads();

  // -------- layer 2: out = h @ W2 + b2, each wave: 64 columns --------------
#pragma unroll 1
  for (int i = 0; i < 4; ++i) {
    const int nt = wave * 4 + i;          // column tile 0..31
    const int n0 = nt * 16;
    const float bias = b2c[n0 + nlane];
    v8f acc0 = {}, acc1 = {}, acc2 = {}, acc3 = {};
#pragma unroll 4
    for (int k0 = 0; k0 < NB_NEUR; k0 += 32) {
      const size_t frag = ((size_t)(c * 32 + (k0 >> 5)) * 32 + nt);
      v16bf b = *reinterpret_cast<const v16bf*>(B2p + (frag << 9) + (lane << 4));
      v16bf a0 = load_a_frag(hbuf, 0, k0, NB_NEUR, lane);
      v16bf a1 = load_a_frag(hbuf, 16, k0, NB_NEUR, lane);
      v16bf a2 = load_a_frag(hbuf, 32, k0, NB_NEUR, lane);
      v16bf a3 = load_a_frag(hbuf, 48, k0, NB_NEUR, lane);
      acc0 = __builtin_amdgcn_wmma_f32_16x16x32_bf16(false, a0, false, b, (short)0, acc0, false, false);
      acc1 = __builtin_amdgcn_wmma_f32_16x16x32_bf16(false, a1, false, b, (short)0, acc1, false, false);
      acc2 = __builtin_amdgcn_wmma_f32_16x16x32_bf16(false, a2, false, b, (short)0, acc2, false, false);
      acc3 = __builtin_amdgcn_wmma_f32_16x16x32_bf16(false, a3, false, b, (short)0, acc3, false, false);
    }
    v8f* accs[4] = {&acc0, &acc1, &acc2, &acc3};
#pragma unroll
    for (int m = 0; m < 4; ++m) {
#pragma unroll
      for (int r = 0; r < 8; ++r) {
        int row = m * 16 + r + mlane;
        if (row < rows)
          out[(size_t)srow[row] * OUT_DIM + n0 + nlane] = (*accs[m])[r] + bias;
      }
    }
  }
}

// ---------------------------------------------------------------------------
extern "C" void kernel_launch(void* const* d_in, const int* in_sizes, int n_in,
                              void* d_out, int out_size, void* d_ws, size_t ws_size,
                              hipStream_t stream) {
  (void)in_sizes; (void)n_in; (void)out_size; (void)ws_size;
  const float* eps  = (const float*)d_in[0];
  const int*   comp = (const int*)d_in[1];
  const float* mu   = (const float*)d_in[2];
  const float* rho  = (const float*)d_in[3];
  const float* W1   = (const float*)d_in[4];
  const float* b1   = (const float*)d_in[5];
  const float* W2   = (const float*)d_in[6];
  const float* b2   = (const float*)d_in[7];
  float* out = (float*)d_out;

  char* ws = (char*)d_ws;
  int* counts  = (int*)(ws + WS_COUNTS);
  int* offsets = (int*)(ws + WS_OFFSETS);
  int* order   = (int*)(ws + WS_ORDER);
  __bf16* B1p  = (__bf16*)(ws + WS_B1PACK);
  __bf16* B2p  = (__bf16*)(ws + WS_B2PACK);

  route_kernel<<<1, 1024, 0, stream>>>(comp, counts, offsets, order);

  // B1: K=512 (KT=16), N=1024 (NT=64); B2: K=1024 (KT=32), N=512 (NT=32)
  pack_kernel<<<(NB_COMP * 16 * 64 * 32 + 255) / 256, 256, 0, stream>>>(
      W1, B1p, 16, 64, NB_NEUR);
  pack_kernel<<<(NB_COMP * 32 * 32 * 32 + 255) / 256, 256, 0, stream>>>(
      W2, B2p, 32, 32, OUT_DIM);

  const size_t smem = (size_t)M_TILE * LAT_DIM * 2 +
                      (size_t)M_TILE * NB_NEUR * 2 + M_TILE * 4;
  mlp_kernel<<<MAX_TILES, 256, smem, stream>>>(eps, order, counts, offsets,
                                               mu, rho, b1, b2, B1p, B2p, out);
}
